// Head_4715874091932
// MI455X (gfx1250) — matrix operand order repaired and checked by reading, compiled, not run on hardware
//
#include <hip/hip_runtime.h>
#include <hip/hip_bf16.h>

// B=4, T=4096, C=1024, H=64 causal attention head, flash-attention style.
// pack_w  : weights -> WMMA-B-fragment-ready bf16 layout
// qkv_proj: fused Q/K/V projection; packed weights async-staged to LDS
//   (double-buffered, block-shared), x streamed through registers, all
//   matmuls via v_wmma_f32_16x16x32_bf16.
// flash_attn: block-cooperative flash attention; K/V tiles double-buffered in
//   LDS via global_load_async_to_lds_b128 (ASYNCcnt) + split barriers; QK^T and
//   PV via WMMA; online softmax with 16-lane xor-shuffle reductions; P matrix
//   transposed D-layout -> A-layout through a per-wave LDS scratch (DScnt).

typedef __attribute__((ext_vector_type(16))) __bf16 v16bf;
typedef __attribute__((ext_vector_type(8)))  __bf16 v8bf;
typedef __attribute__((ext_vector_type(8)))  float  v8f;

#define TT 4096
#define CC 1024
#define HH 64
#define NEGV (-1000000000.0f)

static __device__ __forceinline__ v16bf cat8(v8bf a, v8bf b) {
  return __builtin_shufflevector(a, b, 0,1,2,3,4,5,6,7,8,9,10,11,12,13,14,15);
}

static __device__ __forceinline__ void async_b128(unsigned lds_addr, const void* gaddr) {
  asm volatile("global_load_async_to_lds_b128 %0, %1, off"
               :: "v"(lds_addr), "v"(gaddr) : "memory");
}

// ---------------------------------------------------------------------------
// Kernel 0: pack Wq/Wk/Wv (fp32 [C][H]) into B-fragment-ready bf16 layout:
//   wp[m][ ((kk*4+nt)*32 + lane)*16 + j ] = W[kk*32 + (lane<16?0:8) + (j&7) + (j<8?0:16)][nt*16 + lane%16]
// A 128-wide k-chunk of one matrix is 8192 consecutive elements (16 KB).
// ---------------------------------------------------------------------------
__global__ void pack_w(const float* __restrict__ wq, const float* __restrict__ wk,
                       const float* __restrict__ wv, __bf16* __restrict__ wp) {
  int idx = blockIdx.x * blockDim.x + threadIdx.x;
  if (idx >= 3 * CC * HH) return;
  int m    = idx >> 16;            // 65536 elems per matrix
  int e    = idx & 65535;
  int j    = e & 15;
  int lane = (e >> 4) & 31;
  int nt   = (e >> 9) & 3;
  int kk   = e >> 11;
  int k = kk * 32 + ((lane < 16) ? 0 : 8) + (j & 7) + ((j < 8) ? 0 : 16);
  int n = nt * 16 + (lane & 15);
  const float* src = (m == 0) ? wq : (m == 1) ? wk : wv;
  wp[idx] = (__bf16)src[k * HH + n];
}

// ---------------------------------------------------------------------------
// Kernel 1: fused QKV projection. One wave computes a 16-row tile of Q,K,V.
// Weights are async-staged per 128-wide k-chunk (48 KB, all 3 matrices) into
// double-buffered LDS shared by the block's 8 waves.
// Q,K stored row-major bf16 [B][T][64]; V stored transposed bf16 [B][64][T].
// ---------------------------------------------------------------------------
__global__ void qkv_proj(const float* __restrict__ x, const __bf16* __restrict__ wp,
                         __bf16* __restrict__ qbf, __bf16* __restrict__ kbf,
                         __bf16* __restrict__ vtp) {
  __shared__ alignas(16) __bf16 wl[2][3 * 8192];  // 2 x 48 KB chunk buffers

  const int lane = threadIdx.x & 31;
  const int wid  = threadIdx.x >> 5;
  const int tid  = blockIdx.x * 8 + wid;
  const int b    = tid / (TT / 16);
  const int t0   = (tid % (TT / 16)) * 16;
  const int ln   = lane & 15;
  const int hi   = lane >> 4;

  // Stage one 48 KB weight chunk (3 x 16 KB, each a linear copy): 12 async
  // b128 transfers per thread, 256 threads cover 4 KB per round.
  auto stage = [&](int kc, int bufi) {
    const unsigned idx = threadIdx.x;
#pragma unroll
    for (int m = 0; m < 3; ++m)
#pragma unroll
      for (int t = 0; t < 4; ++t) {
        const unsigned eo = (t * 256 + idx) * 8;  // element offset within 16 KB region
        const __bf16* g = wp + m * (CC * HH) + kc * 8192 + eo;
        unsigned l = (unsigned)(size_t)(&wl[bufi][0]) + (m * 8192 + eo) * 2;
        async_b128(l, g);
      }
  };

  v8f accq[4], acck[4], accv[4];
#pragma unroll
  for (int i = 0; i < 4; ++i) { accq[i] = 0; acck[i] = 0; accv[i] = 0; }

  const float* xrow = x + ((size_t)b * TT + t0 + ln) * CC;

  stage(0, 0);
  asm volatile("s_wait_asynccnt 0x0" ::: "memory");
  __syncthreads();

  for (int kc = 0; kc < CC / 128; ++kc) {   // 8 chunks of 128 k-values
    const int cur = kc & 1;
    if (kc + 1 < CC / 128) stage(kc + 1, cur ^ 1);  // overlap next-chunk copy

#pragma unroll
    for (int kk2 = 0; kk2 < 4; ++kk2) {     // 4 k-steps of 32 within the chunk
      const int kk = kc * 4 + kk2;
      // A fragment of x: lane<16 -> K={0..7,16..23}, lane>=16 -> K={8..15,24..31}
      const float* xp = xrow + kk * 32 + (hi ? 8 : 0);
      if (kk + 1 < CC / 32) __builtin_prefetch(xrow + (kk + 1) * 32, 0, 0);
      float4 f0 = *(const float4*)(xp);
      float4 f1 = *(const float4*)(xp + 4);
      float4 f2 = *(const float4*)(xp + 16);
      float4 f3 = *(const float4*)(xp + 20);
      v16bf ax;
      ax[0]=(__bf16)f0.x; ax[1]=(__bf16)f0.y; ax[2]=(__bf16)f0.z; ax[3]=(__bf16)f0.w;
      ax[4]=(__bf16)f1.x; ax[5]=(__bf16)f1.y; ax[6]=(__bf16)f1.z; ax[7]=(__bf16)f1.w;
      ax[8]=(__bf16)f2.x; ax[9]=(__bf16)f2.y; ax[10]=(__bf16)f2.z; ax[11]=(__bf16)f2.w;
      ax[12]=(__bf16)f3.x; ax[13]=(__bf16)f3.y; ax[14]=(__bf16)f3.z; ax[15]=(__bf16)f3.w;

#pragma unroll
      for (int nt = 0; nt < 4; ++nt) {
        const int fo = ((kk2 * 4 + nt) * 32 + lane) * 16;
        const __bf16* pq = &wl[cur][0 * 8192 + fo];
        const __bf16* pk = &wl[cur][1 * 8192 + fo];
        const __bf16* pv = &wl[cur][2 * 8192 + fo];
        v16bf bq = cat8(*(const v8bf*)pq, *(const v8bf*)(pq + 8));
        v16bf bk = cat8(*(const v8bf*)pk, *(const v8bf*)(pk + 8));
        v16bf bv = cat8(*(const v8bf*)pv, *(const v8bf*)(pv + 8));
        accq[nt] = __builtin_amdgcn_wmma_f32_16x16x32_bf16(false, ax, false, bq,
                                                           (short)0, accq[nt], false, false);
        acck[nt] = __builtin_amdgcn_wmma_f32_16x16x32_bf16(false, ax, false, bk,
                                                           (short)0, acck[nt], false, false);
        accv[nt] = __builtin_amdgcn_wmma_f32_16x16x32_bf16(false, ax, false, bv,
                                                           (short)0, accv[nt], false, false);
      }
    }

    asm volatile("s_wait_asynccnt 0x0" ::: "memory");
    __syncthreads();
  }

  // Store. D-layout: lane%16 = column (h within tile), VGPR r = row r (+8 for hi).
#pragma unroll
  for (int nt = 0; nt < 4; ++nt) {
    const int h = nt * 16 + ln;
#pragma unroll
    for (int r = 0; r < 8; ++r) {
      const int t = t0 + r + hi * 8;
      qbf[((size_t)b * TT + t) * HH + h] = (__bf16)accq[nt][r];
      kbf[((size_t)b * TT + t) * HH + h] = (__bf16)acck[nt][r];
    }
    v8bf pv;
#pragma unroll
    for (int r = 0; r < 8; ++r) pv[r] = (__bf16)accv[nt][r];
    *(v8bf*)(vtp + ((size_t)b * HH + h) * TT + t0 + hi * 8) = pv;
  }
}

// ---------------------------------------------------------------------------
// Kernel 2: block-cooperative flash attention with async double-buffered K/V.
// Block = 8 waves = 8 consecutive 16-query tiles (never crosses a batch).
// ---------------------------------------------------------------------------
__global__ void flash_attn(const __bf16* __restrict__ qbf, const __bf16* __restrict__ kbf,
                           const __bf16* __restrict__ vtp, float* __restrict__ out) {
  __shared__ alignas(16) __bf16 kbuf[2][32 * HH];   // [key][h]   row-major, 4 KB each
  __shared__ alignas(16) __bf16 vbuf[2][HH * 32];   // [h][key]   row-major, 4 KB each
  __shared__ alignas(16) __bf16 psc[8][16 * 32];    // per-wave P transpose scratch

  const int lane = threadIdx.x & 31;
  const int wid  = threadIdx.x >> 5;
  const int tid  = blockIdx.x * 8 + wid;
  const int b    = tid / (TT / 16);
  const int t0   = (tid % (TT / 16)) * 16;
  const int ln   = lane & 15;
  const int hi   = lane >> 4;
  __bf16* pw = &psc[wid][0];

  const int base_t0 = ((blockIdx.x * 8) % (TT / 16)) * 16;  // first query row of block
  const int nsteps  = (base_t0 + 127) / 32 + 1;             // causal bound for whole block
  const int my_smax = t0 + 15;                              // this wave's last needed key

  // Cooperative async stage of one 32-key tile (K: 4 KB contiguous; V^T: 64x64B rows).
  auto stage = [&](int s0, int bufi) {
    const unsigned idx = threadIdx.x;  // 0..255, each lane moves 16 bytes per matrix
    {
      const __bf16* g = kbf + ((size_t)b * TT + s0) * HH + idx * 8;
      unsigned l = (unsigned)(size_t)(&kbuf[bufi][0]) + idx * 16;
      async_b128(l, g);
    }
    {
      const unsigned h = idx >> 2, c = idx & 3;
      const __bf16* g = vtp + ((size_t)b * HH + h) * TT + s0 + c * 8;
      unsigned l = (unsigned)(size_t)(&vbuf[bufi][0]) + (h * 32 + c * 8) * 2;
      async_b128(l, g);
    }
  };

  // Q A-fragments (K-dim = H = 64 -> two 16x16x32 chunks), kept in registers.
  const __bf16* qrow = qbf + ((size_t)b * TT + t0 + ln) * HH;
  v16bf qf[2];
#pragma unroll
  for (int c = 0; c < 2; ++c) {
    const __bf16* p = qrow + c * 32 + (hi ? 8 : 0);
    qf[c] = cat8(*(const v8bf*)p, *(const v8bf*)(p + 16));
  }

  v8f acc[4];
#pragma unroll
  for (int i = 0; i < 4; ++i) acc[i] = 0;
  float m[8], l[8], alpha[8];
#pragma unroll
  for (int r = 0; r < 8; ++r) { m[r] = -3.0e30f; l[r] = 0.0f; }
  const float scale = 0.125f;  // 64^-0.5

  stage(0, 0);
  asm volatile("s_wait_asynccnt 0x0" ::: "memory");
  __syncthreads();

  for (int i = 0; i < nsteps; ++i) {
    const int s0  = i * 32;
    const int cur = i & 1;
    if (i + 1 < nsteps) stage((i + 1) * 32, cur ^ 1);  // overlap next-tile copy

    if (s0 <= my_smax) {  // wave-uniform causal skip; EXEC stays all-ones inside
      // S = Q K^T : two 16x16 D-fragments (keys s0..s0+15 and s0+16..s0+31)
      v8f slo = 0, shi2 = 0;
#pragma unroll
      for (int c = 0; c < 2; ++c) {
        const __bf16* kp0 = &kbuf[cur][ln * HH + c * 32 + (hi ? 8 : 0)];
        const __bf16* kp1 = &kbuf[cur][(16 + ln) * HH + c * 32 + (hi ? 8 : 0)];
        v16bf kf0 = cat8(*(const v8bf*)kp0, *(const v8bf*)(kp0 + 16));
        v16bf kf1 = cat8(*(const v8bf*)kp1, *(const v8bf*)(kp1 + 16));
        slo  = __builtin_amdgcn_wmma_f32_16x16x32_bf16(false, qf[c], false, kf0,
                                                       (short)0, slo, false, false);
        shi2 = __builtin_amdgcn_wmma_f32_16x16x32_bf16(false, qf[c], false, kf1,
                                                       (short)0, shi2, false, false);
      }

      // Causal mask + online softmax (row r -> query t0+r+hi*8; col = lane%16).
#pragma unroll
      for (int r = 0; r < 8; ++r) {
        const int row = t0 + r + hi * 8;
        float a  = slo[r]  * scale; if (s0 + ln      > row) a  = NEGV;
        float b2 = shi2[r] * scale; if (s0 + 16 + ln > row) b2 = NEGV;
        float v = fmaxf(a, b2);
#pragma unroll
        for (int off = 1; off < 16; off <<= 1) v = fmaxf(v, __shfl_xor(v, off, 32));
        const float mn = fmaxf(m[r], v);
        const float plo = __expf(a - mn);
        const float phi = __expf(b2 - mn);
        float rs = plo + phi;
#pragma unroll
        for (int off = 1; off < 16; off <<= 1) rs += __shfl_xor(rs, off, 32);
        alpha[r] = __expf(m[r] - mn);
        l[r] = l[r] * alpha[r] + rs;
        m[r] = mn;
        pw[(r + hi * 8) * 32 + ln]      = (__bf16)plo;   // D-layout -> LDS
        pw[(r + hi * 8) * 32 + ln + 16] = (__bf16)phi;
      }
#pragma unroll
      for (int nt = 0; nt < 4; ++nt)
#pragma unroll
        for (int r = 0; r < 8; ++r) acc[nt][r] *= alpha[r];

      // Same-wave DS ops are in-order; wait for stores, reload P in A-layout.
      asm volatile("s_wait_dscnt 0" ::: "memory");
      const __bf16* pr = pw + ln * 32 + (hi ? 8 : 0);
      v16bf pf = cat8(*(const v8bf*)pr, *(const v8bf*)(pr + 16));

      // O += P V
#pragma unroll
      for (int nt = 0; nt < 4; ++nt) {
        const __bf16* vp = &vbuf[cur][(nt * 16 + ln) * 32 + (hi ? 8 : 0)];
        v16bf vf = cat8(*(const v8bf*)vp, *(const v8bf*)(vp + 16));
        acc[nt] = __builtin_amdgcn_wmma_f32_16x16x32_bf16(false, pf, false, vf,
                                                          (short)0, acc[nt], false, false);
      }
    }

    // Own async copies must land before anyone reads the other buffer.
    asm volatile("s_wait_asynccnt 0x0" ::: "memory");
    __syncthreads();
  }

  // Epilogue: out = acc / l, fp32 row-major [B][T][64].
#pragma unroll
  for (int nt = 0; nt < 4; ++nt) {
#pragma unroll
    for (int r = 0; r < 8; ++r) {
      const int t = t0 + r + hi * 8;
      out[((size_t)b * TT + t) * HH + nt * 16 + ln] = acc[nt][r] / l[r];
    }
  }
}

// ---------------------------------------------------------------------------
extern "C" void kernel_launch(void* const* d_in, const int* in_sizes, int n_in,
                              void* d_out, int out_size, void* d_ws, size_t ws_size,
                              hipStream_t stream) {
  const float* x  = (const float*)d_in[0];
  const float* Wk = (const float*)d_in[1];
  const float* Wq = (const float*)d_in[2];
  const float* Wv = (const float*)d_in[3];

  char* ws = (char*)d_ws;
  __bf16* wp  = (__bf16*)ws;                          // 3*1024*64*2 = 384 KB
  __bf16* qbf = (__bf16*)(ws + 393216);               // 2 MB
  __bf16* kbf = (__bf16*)(ws + 393216 + 2097152);     // 2 MB
  __bf16* vtp = (__bf16*)(ws + 393216 + 2 * 2097152); // 2 MB (transposed V)

  pack_w<<<768, 256, 0, stream>>>(Wq, Wk, Wv, wp);
  qkv_proj<<<128, 256, 0, stream>>>(x, wp, qbf, kbf, vtp);
  flash_attn<<<128, 256, 0, stream>>>(qbf, kbf, vtp, (float*)d_out);
}
